// Tnn_Bin_Conv2d_35287451304617
// MI455X (gfx1250) — 8-bit matrix operand order repaired and checked by reading, compiled, not run on hardware
//
#include <hip/hip_runtime.h>
#include <stdint.h>

typedef __attribute__((ext_vector_type(8))) int v8i;

#define BSZ   32
#define CIN   256
#define COUT  256
#define Hc    56
#define Wc    56
#define HW    (Hc*Wc)
#define MTOT  (BSZ*HW)
#define KTOT  (9*CIN)
#define KSTEPS (KTOT/64)
#define MBLK  128
#define NBLK  128
#define NMB   (MTOT/MBLK)

#define WS_XQ     0u
#define WS_BT     25690112u
#define WS_ALPHA  (WS_BT + 589824u)
#define WS_PSUM   (WS_ALPHA + 1024u)
#define WS_PSQ    (WS_PSUM + (unsigned)(NMB*COUT*4))
#define WS_SCALE  (WS_PSQ  + (unsigned)(NMB*COUT*4))
#define WS_SHIFT  (WS_SCALE + 1024u)

__device__ __forceinline__ int sgn8(float f) { return (f > 0.f) - (f < 0.f); }

__device__ __forceinline__ void async_ld16(unsigned lds_off, const void* gaddr) {
  asm volatile("global_load_async_to_lds_b128 %0, %1, off"
               :: "v"(lds_off), "v"(gaddr));
}
__device__ __forceinline__ void wait_async0() {
  asm volatile("s_wait_asynccnt 0x0" ::: "memory");
}

__global__ void k_quant_x(const float* __restrict__ x, int8_t* __restrict__ xq) {
  int idx = blockIdx.x * 256 + threadIdx.x;
  int p  = idx >> 6;
  int c0 = (idx & 63) << 2;
  int b  = p / HW, hw = p - b * HW;
  const float* src = x + ((size_t)b * CIN + c0) * HW + hw;
  int s0 = sgn8(src[0]);
  int s1 = sgn8(src[HW]);
  int s2 = sgn8(src[2 * HW]);
  int s3 = sgn8(src[3 * HW]);
  unsigned pk = (unsigned)(s0 & 255) | ((unsigned)(s1 & 255) << 8) |
                ((unsigned)(s2 & 255) << 16) | ((unsigned)(s3 & 255) << 24);
  *(unsigned*)(xq + (size_t)p * CIN + c0) = pk;
}

__global__ void k_quant_w(const float* __restrict__ w, int8_t* __restrict__ bt,
                          float* __restrict__ alpha) {
  int o = blockIdx.x;
  int t = threadIdx.x;
  __shared__ float red[256];
  const float* wrow = w + ((size_t)o * CIN + t) * 9;
  float tap[9], m[9];
#pragma unroll
  for (int j = 0; j < 9; j++) tap[j] = wrow[j];
  for (int j = 0; j < 9; j++) {
    red[t] = tap[j]; __syncthreads();
    for (int s = 128; s > 0; s >>= 1) { if (t < s) red[t] += red[t + s]; __syncthreads(); }
    m[j] = red[0] * (1.f / CIN); __syncthreads();
  }
  float asum = 0.f;
#pragma unroll
  for (int j = 0; j < 9; j++) {
    float wc = tap[j] - m[j];
    wc = fminf(1.f, fmaxf(-1.f, wc));
    asum += fabsf(wc);
    bt[(size_t)o * KTOT + j * CIN + t] = (int8_t)sgn8(wc);
  }
  red[t] = asum; __syncthreads();
  for (int s = 128; s > 0; s >>= 1) { if (t < s) red[t] += red[t + s]; __syncthreads(); }
  if (t == 0) alpha[o] = red[0] * (1.f / KTOT);
}

__global__ void __launch_bounds__(256, 1)
k_gemm(const int8_t* __restrict__ xq, const int8_t* __restrict__ bt,
       const float* __restrict__ alpha, const float* __restrict__ bias,
       float* __restrict__ out, float* __restrict__ psum,
       float* __restrict__ psq) {
  __shared__ __align__(16) char smem[32768];
  int8_t* aS = (int8_t*)smem;
  const unsigned smemBase = (unsigned)(uintptr_t)smem;

  const int tid  = threadIdx.x;
  const int lane = tid & 31;
  const int wave = tid >> 5;
  const int wm = wave & 3, wn = wave >> 2;
  const int half = lane >> 4;
  const int l16  = lane & 15;
  const int m0 = blockIdx.x * MBLK;
  const int n0 = blockIdx.y * NBLK;

  int hA[2], wA[2], shA[2];
  unsigned offA[2], offB[2];
#pragma unroll
  for (int p = 0; p < 2; p++) {
    int chunk = tid + p * 256;
    int r = chunk >> 2, qd = chunk & 3;
    int m = m0 + r;
    int b = m / HW, hw = m - b * HW;
    hA[p] = hw / Wc; wA[p] = hw - hA[p] * Wc;
    offA[p] = (unsigned)(b * HW + hw) * CIN + qd * 16;
    shA[p] = r * 64 + qd * 16;
    offB[p] = (unsigned)(n0 + r) * KTOT + qd * 16;
  }

  v8i acc[2][4];
  const v8i vzero = {0, 0, 0, 0, 0, 0, 0, 0};
#pragma unroll
  for (int i = 0; i < 2; i++)
#pragma unroll
    for (int j = 0; j < 4; j++) acc[i][j] = vzero;

  union Frag { v8i v; int4 q[2]; unsigned long long d[4]; };
  const int4 z4 = make_int4(0, 0, 0, 0);

  auto issue = [&](int kc, int buf) {
    int tap = kc >> 2;
    int ci0 = (kc & 3) * 64;
    int t3 = tap / 3;
    int dh = t3 - 1, dw = tap - t3 * 3 - 1;
    int doff = (dh * Wc + dw) * CIN + ci0;
#pragma unroll
    for (int p = 0; p < 2; p++) {
      int hs = hA[p] + dh, ws = wA[p] + dw;
      bool v = ((unsigned)hs < (unsigned)Hc) && ((unsigned)ws < (unsigned)Wc);
      int off = v ? doff : ci0;
      async_ld16(smemBase + buf * 8192 + shA[p], xq + (offA[p] + (unsigned)off));
      async_ld16(smemBase + 16384 + buf * 8192 + shA[p], bt + (offB[p] + (unsigned)(kc * 64)));
    }
  };
  auto fix = [&](int kc, int buf) {
    int tap = kc >> 2;
    int t3 = tap / 3;
    int dh = t3 - 1, dw = tap - t3 * 3 - 1;
#pragma unroll
    for (int p = 0; p < 2; p++) {
      int hs = hA[p] + dh, ws = wA[p] + dw;
      if (!(((unsigned)hs < (unsigned)Hc) && ((unsigned)ws < (unsigned)Wc)))
        *(int4*)&aS[buf * 8192 + shA[p]] = z4;
    }
  };

  issue(0, 0);
  wait_async0();
  fix(0, 0);
  __syncthreads();

  for (int kc = 0; kc < KSTEPS; kc++) {
    const int cur = kc & 1;
    if (kc + 1 < KSTEPS) issue(kc + 1, 1 - cur);

    Frag afr[2];
#pragma unroll
    for (int mi = 0; mi < 2; mi++) {
      const int8_t* ap = &aS[cur * 8192 + (wm * 32 + mi * 16 + l16) * 64 + half * 16];
      afr[mi].d[0] = *(const unsigned long long*)(ap);
      afr[mi].d[1] = *(const unsigned long long*)(ap + 8);
      afr[mi].d[2] = *(const unsigned long long*)(ap + 32);
      afr[mi].d[3] = *(const unsigned long long*)(ap + 40);
    }
    Frag bfr[4];
#pragma unroll
    for (int ni = 0; ni < 4; ni++) {
      const int8_t* bp = &aS[16384 + cur * 8192 + (wn * 64 + ni * 16 + l16) * 64 + half * 16];
      bfr[ni].q[0] = *(const int4*)(bp);
      bfr[ni].q[1] = *(const int4*)(bp + 32);
    }
#pragma unroll
    for (int ni = 0; ni < 4; ni++) {
      acc[0][ni] = __builtin_amdgcn_wmma_i32_16x16x64_iu8(
          true, afr[0].v, true, bfr[ni].v, acc[0][ni], false, false);
      acc[1][ni] = __builtin_amdgcn_wmma_i32_16x16x64_iu8(
          true, afr[1].v, true, bfr[ni].v, acc[1][ni], false, false);
    }

    if (kc + 1 < KSTEPS) { wait_async0(); fix(kc + 1, 1 - cur); }
    __syncthreads();
  }

  float* ly = (float*)smem;
  float* ps = (float*)(smem + 17024);
  float* pq = (float*)(smem + 17024 + 4096);

  const int mloc = tid & 127;
  const int colb = tid >> 7;
  const int me = m0 + mloc;
  const int bb = me / HW;
  const int hwme = me - bb * HW;
  float* outb = out + (size_t)bb * COUT * HW + hwme;

  float ls[4] = {0.f, 0.f, 0.f, 0.f}, lq[4] = {0.f, 0.f, 0.f, 0.f};
#pragma unroll
  for (int ni = 0; ni < 4; ni++) {
    const int n = n0 + wn * 64 + ni * 16 + l16;
    const float al = alpha[n], bi = bias[n];
#pragma unroll
    for (int mi = 0; mi < 2; mi++) {
#pragma unroll
      for (int g = 0; g < 8; g++) {
        int ml = wm * 32 + mi * 16 + half * 8 + g;
        float y = al * (float)acc[mi][ni][g] + bi;
        ls[ni] += y; lq[ni] += y * y;
        ly[ml * 33 + wn * 16 + l16] = y;
      }
    }
    __syncthreads();
#pragma unroll
    for (int cp = 0; cp < 16; cp++) {
      int col = colb + cp * 2;
      int wn2 = col >> 4;
      int c = n0 + wn2 * 64 + ni * 16 + (col & 15);
      outb[(size_t)c * HW] = ly[mloc * 33 + col];
    }
    __syncthreads();
  }

#pragma unroll
  for (int ni = 0; ni < 4; ni++) {
    ps[(wave * 32 + lane) * 4 + ni] = ls[ni];
    pq[(wave * 32 + lane) * 4 + ni] = lq[ni];
  }
  __syncthreads();
  if (tid < NBLK) {
    int wn_ = tid >> 6, rem = tid & 63, ni = rem >> 4, col = rem & 15;
    float s = 0.f, q = 0.f;
#pragma unroll
    for (int wm_ = 0; wm_ < 4; wm_++) {
      int wv = wn_ * 4 + wm_;
      s += ps[(wv * 32 + col) * 4 + ni] + ps[(wv * 32 + col + 16) * 4 + ni];
      q += pq[(wv * 32 + col) * 4 + ni] + pq[(wv * 32 + col + 16) * 4 + ni];
    }
    int ch = n0 + tid;
    psum[(size_t)blockIdx.x * COUT + ch] = s;
    psq [(size_t)blockIdx.x * COUT + ch] = q;
  }
}

__global__ void k_stats(const float* __restrict__ psum, const float* __restrict__ psq,
                        const float* __restrict__ gamma, const float* __restrict__ beta,
                        float* __restrict__ scale, float* __restrict__ shift) {
  int c = blockIdx.x, t = threadIdx.x;
  __shared__ float rs[256], rq[256];
  float s = 0.f, q = 0.f;
  for (int i = t; i < NMB; i += 256) { s += psum[(size_t)i * COUT + c]; q += psq[(size_t)i * COUT + c]; }
  rs[t] = s; rq[t] = q; __syncthreads();
  for (int st = 128; st > 0; st >>= 1) {
    if (t < st) { rs[t] += rs[t + st]; rq[t] += rq[t + st]; }
    __syncthreads();
  }
  if (t == 0) {
    float mean = rs[0] * (1.f / MTOT);
    float var  = rq[0] * (1.f / MTOT) - mean * mean;
    float sc   = gamma[c] * rsqrtf(var + 1e-5f);
    scale[c] = sc;
    shift[c] = beta[c] - mean * sc;
  }
}

__global__ void k_final(float* __restrict__ out, const float* __restrict__ scale,
                        const float* __restrict__ shift) {
  size_t i4 = (size_t)blockIdx.x * 256 + threadIdx.x;
  float4 v = ((float4*)out)[i4];
  int c = (int)((i4 * 4 / HW) % COUT);
  float sc = scale[c], sh = shift[c];
  v.x = fmaxf(v.x * sc + sh, 0.f);
  v.y = fmaxf(v.y * sc + sh, 0.f);
  v.z = fmaxf(v.z * sc + sh, 0.f);
  v.w = fmaxf(v.w * sc + sh, 0.f);
  ((float4*)out)[i4] = v;
}

extern "C" void kernel_launch(void* const* d_in, const int* in_sizes, int n_in,
                              void* d_out, int out_size, void* d_ws, size_t ws_size,
                              hipStream_t stream) {
  const float* x      = (const float*)d_in[0];
  const float* weight = (const float*)d_in[1];
  const float* bias   = (const float*)d_in[2];
  const float* gamma  = (const float*)d_in[3];
  const float* beta   = (const float*)d_in[4];
  float* out = (float*)d_out;

  uint8_t* ws = (uint8_t*)d_ws;
  int8_t* xq    = (int8_t*)(ws + WS_XQ);
  int8_t* bt    = (int8_t*)(ws + WS_BT);
  float*  alpha = (float*) (ws + WS_ALPHA);
  float*  psum  = (float*) (ws + WS_PSUM);
  float*  psq   = (float*) (ws + WS_PSQ);
  float*  scale = (float*) (ws + WS_SCALE);
  float*  shift = (float*) (ws + WS_SHIFT);

  k_quant_x<<<MTOT * (CIN / 4) / 256, 256, 0, stream>>>(x, xq);
  k_quant_w<<<COUT, 256, 0, stream>>>(weight, bt, alpha);
  dim3 gg(MTOT / MBLK, COUT / NBLK);
  k_gemm<<<gg, 256, 0, stream>>>(xq, bt, alpha, bias, out, psum, psq);
  k_stats<<<COUT, 256, 0, stream>>>(psum, psq, gamma, beta, scale, shift);
  k_final<<<(BSZ * COUT * HW / 4) / 256, 256, 0, stream>>>(out, scale, shift);
}
